// MultiHeadAttentionBlock_57569741635752
// MI455X (gfx1250) — matrix operand order repaired and checked by reading, not compiled, not run on hardware
//
#include <hip/hip_runtime.h>

// Problem constants (match reference)
#define B_  2
#define S_  2048
#define D_  768
#define H_  12
#define DK_ 64
#define M_  (B_ * S_)   // 4096 rows for all projection GEMMs

typedef __attribute__((ext_vector_type(16))) _Float16 v16h;
typedef __attribute__((ext_vector_type(8)))  _Float16 v8h;
typedef __attribute__((ext_vector_type(4)))  _Float16 v4h;
typedef __attribute__((ext_vector_type(8)))  float    v8f;
typedef __attribute__((ext_vector_type(4)))  float    v4f;

// ---------------------------------------------------------------------------
// WMMA helpers (gfx1250 wave32, 16x16x32 f16 -> f32)
// ---------------------------------------------------------------------------
__device__ __forceinline__ v8f wmma_f16(v16h a, v16h b, v8f c) {
  // (neg_a, A, neg_b, B, c_mod, C, reuse_a, reuse_b)
  return __builtin_amdgcn_wmma_f32_16x16x32_f16(false, a, false, b, (short)0, c,
                                                false, false);
}

// Build the 16x32 f16 A-operand for one lane.
// Layout used here (B matches): lane group g = lane>>4, row M = lane&15.
//   halves 0..15 -> K = k0 + g*16 + 0..15 (contiguous)
// 'rowk' points at (row base + k0); two contiguous 16B loads per lane.
__device__ __forceinline__ v16h make_a(const _Float16* rowk, int g) {
  v8h lo = *(const v8h*)(rowk + g * 16);
  v8h hi = *(const v8h*)(rowk + g * 16 + 8);
  return __builtin_shufflevector(lo, hi, 0, 1, 2, 3, 4, 5, 6, 7, 8, 9, 10, 11,
                                 12, 13, 14, 15);
}

// ---------------------------------------------------------------------------
// f32 -> f16 cast (x4 vectorized)
// ---------------------------------------------------------------------------
__global__ void cast_f32_to_f16(const float* __restrict__ src,
                                _Float16* __restrict__ dst, int n4) {
  int i = blockIdx.x * blockDim.x + threadIdx.x;
  if (i < n4) {
    v4f x = ((const v4f*)src)[i];
    v4h y;
    y[0] = (_Float16)x[0];
    y[1] = (_Float16)x[1];
    y[2] = (_Float16)x[2];
    y[3] = (_Float16)x[3];
    ((v4h*)dst)[i] = y;
  }
}

// ---------------------------------------------------------------------------
// WMMA GEMM:  Y = X[M,K] @ W[N,K]^T + bias, per-wave 32x64 tile
// (2 A-operands x 4 B-operands -> 8 WMMA per 32-wide k-step).
// MODE 0: store f16 head-split [B,H,S,DK]   (Q / K projections; scale folded)
// MODE 1: store f16 transposed [B,H,DK,S]   (V projection, for P@V B-operand)
// MODE 2: store f32 row-major  [M,N]        (final output projection)
// ---------------------------------------------------------------------------
template <int MODE>
__global__ __launch_bounds__(256) void gemm_wmma16(
    const _Float16* __restrict__ X,    // [M,K] row-major f16
    const _Float16* __restrict__ W,    // [N,K] row-major f16
    const float* __restrict__ bias,    // [N] f32
    void* __restrict__ Yv, int M, int N, int K, float scale) {
  const int lane = threadIdx.x & 31;
  const int wave = threadIdx.x >> 5;
  const int g = lane >> 4;
  const int ln = lane & 15;

  const int ntn = N >> 6;  // tiles of 64 columns
  const int w = blockIdx.x * 8 + wave;
  const int m0 = (w / ntn) * 32;
  const int n0 = (w % ntn) * 64;

  const _Float16* xrow0 = X + (size_t)(m0 + ln) * K;
  const _Float16* xrow1 = X + (size_t)(m0 + 16 + ln) * K;

  v8f acc[8];  // [u][t]: u = m-subtile 0..1, t = n-subtile 0..3
  v8f z = {};
#pragma unroll
  for (int t = 0; t < 8; ++t) acc[t] = z;

  for (int k0 = 0; k0 < K; k0 += 32) {
    v16h a0 = make_a(xrow0 + k0, g);
    v16h a1 = make_a(xrow1 + k0, g);
    __builtin_prefetch(xrow0 + k0 + 128, 0, 0);
    __builtin_prefetch(xrow1 + k0 + 128, 0, 0);
#pragma unroll
    for (int t = 0; t < 4; ++t) {
      // B operand: lane = column n, halves = K = k0 + g*16 + 0..15 contiguous
      const _Float16* wp = W + (size_t)(n0 + t * 16 + ln) * K + k0 + g * 16;
      v16h b = *(const v16h*)wp;
      acc[t] = wmma_f16(a0, b, acc[t]);
      acc[4 + t] = wmma_f16(a1, b, acc[4 + t]);
    }
  }

  // C/D layout: VGPR r, lane (g,ln) holds [M = m0 + u*16 + r + g*8,
  //                                        N = n0 + t*16 + ln]
#pragma unroll
  for (int u = 0; u < 2; ++u) {
#pragma unroll
    for (int t = 0; t < 4; ++t) {
      const int n = n0 + t * 16 + ln;
      const float bv = bias[n];
#pragma unroll
      for (int r = 0; r < 8; ++r) {
        const int m = m0 + u * 16 + r + g * 8;
        const float val = (acc[u * 4 + t][r] + bv) * scale;
        if (MODE == 0) {
          const int b = m >> 11, s = m & (S_ - 1);
          const int h = n >> 6, d = n & 63;
          ((_Float16*)Yv)[(((size_t)(b * H_ + h) * S_ + s) << 6) + d] =
              (_Float16)val;
        } else if (MODE == 1) {
          const int b = m >> 11, s = m & (S_ - 1);
          const int h = n >> 6, d = n & 63;
          ((_Float16*)Yv)[((size_t)(b * H_ + h) * DK_ + d) * S_ + s] =
              (_Float16)val;
        } else {
          ((float*)Yv)[(size_t)m * N + n] = val;
        }
      }
    }
  }
}

// ---------------------------------------------------------------------------
// Causal flash attention: one wave owns a 16-query tile of one (b,h).
// Qp,Kp: [B,H,S,DK] f16 (Q pre-scaled by 1/sqrt(DK)); Vt: [B,H,DK,S] f16.
// Softmax strategy: mask raw scores in C-layout, round-trip through LDS into
// WMMA A-layout (needed for P@V anyway); in A-layout a row lives in lane m
// and lane m^16, so row max/sum cost ONE shfl_xor each instead of 32
// bpermutes. exp() output lands directly in the A operand.
// ---------------------------------------------------------------------------
__global__ __launch_bounds__(256) void attn_flash(
    const _Float16* __restrict__ Qp, const _Float16* __restrict__ Kp,
    const _Float16* __restrict__ Vt, _Float16* __restrict__ O16) {
  // Per-wave 16x32 score tile, row stride 40 halves (80B) to spread banks.
  __shared__ __align__(32) _Float16 plds[8][16 * 40];

  const int lane = threadIdx.x & 31;
  const int wave = threadIdx.x >> 5;
  const int g = lane >> 4;
  const int ln = lane & 15;

  const int qtiles = S_ / 16;
  const int w = blockIdx.x * 8 + wave;
  const int bh = w / qtiles;
  const int q0 = (w % qtiles) * 16;

  const _Float16* Qb = Qp + (size_t)bh * S_ * DK_;
  const _Float16* Kb = Kp + (size_t)bh * S_ * DK_;
  const _Float16* Vb = Vt + (size_t)bh * DK_ * S_;

  // Q A-operands for both 32-wide K-steps of the d=64 contraction (kept live).
  const _Float16* qrow = Qb + (size_t)(q0 + ln) * DK_;
  const v16h aq0 = make_a(qrow, g);
  const v16h aq1 = make_a(qrow + 32, g);

  // Per-lane online-softmax state for row m = ln (duplicated in both groups).
  float mrow = -1e30f;
  float lrow = 0.0f;
  v8f acc[4];
  v8f z = {};
#pragma unroll
  for (int t = 0; t < 4; ++t) acc[t] = z;

  _Float16* pw = &plds[wave][0];

  for (int kc = 0; kc < q0 + 16; kc += 32) {
    // ---- scores: two 16x16 key sub-tiles, contraction over d (2 WMMA each)
    const int key0 = kc + ln;
    const int key1 = kc + 16 + ln;
    const int k0c = key0 > S_ - 1 ? S_ - 1 : key0;  // clamped (masked anyway)
    const int k1c = key1 > S_ - 1 ? S_ - 1 : key1;
    const _Float16* k0p = Kb + (size_t)k0c * DK_ + g * 16;
    const _Float16* k1p = Kb + (size_t)k1c * DK_ + g * 16;
    v8f s0 = z, s1 = z;
    s0 = wmma_f16(aq0, *(const v16h*)(k0p), s0);
    s0 = wmma_f16(aq1, *(const v16h*)(k0p + 32), s0);
    s1 = wmma_f16(aq0, *(const v16h*)(k1p), s1);
    s1 = wmma_f16(aq1, *(const v16h*)(k1p + 32), s1);

    // ---- causal mask in C-layout, store masked f16 scores to LDS
#pragma unroll
    for (int r = 0; r < 8; ++r) {
      const int row = q0 + r + g * 8;
      const float a0 = (key0 > row) ? -30000.0f : s0[r];
      const float a1 = (key1 > row) ? -30000.0f : s1[r];
      const int rr = r + g * 8;
      pw[rr * 40 + ln] = (_Float16)a0;
      pw[rr * 40 + 16 + ln] = (_Float16)a1;
    }
    asm volatile("s_wait_dscnt 0" ::: "memory");

    // ---- read row m=ln back in A-operand layout (two b128 LDS reads)
    const v16h sc = make_a(pw + ln * 40, g);

    // ---- row stats: per-lane reduce + one cross-group shuffle each
    float x[16];
    float lmax = -3.0e38f;
#pragma unroll
    for (int h = 0; h < 16; ++h) {
      x[h] = (float)sc[h];
      lmax = fmaxf(lmax, x[h]);
    }
    lmax = fmaxf(lmax, __shfl_xor(lmax, 16, 32));
    const float nm = fmaxf(mrow, lmax);
    const float alpha = __expf(mrow - nm);
    mrow = nm;

    float lsum = 0.0f;
    v16h ap;  // P in A-operand layout, built in-register
#pragma unroll
    for (int h = 0; h < 16; ++h) {
      const float p = __expf(x[h] - nm);
      lsum += p;
      ap[h] = (_Float16)p;
    }
    lsum += __shfl_xor(lsum, 16, 32);
    lrow = lrow * alpha + lsum;

    // ---- rescale accumulators: broadcast alpha back into C-layout rows
    float alphaC[8];
#pragma unroll
    for (int r = 0; r < 8; ++r) alphaC[r] = __shfl(alpha, r + g * 8, 32);
#pragma unroll
    for (int t = 0; t < 4; ++t)
#pragma unroll
      for (int r = 0; r < 8; ++r) acc[t][r] *= alphaC[r];

    // ---- P @ V: B operand contiguous from transposed V
    int ks = kc + g * 16;
    if (ks > S_ - 16) ks = S_ - 16;  // clamped (masked keys have P==0)
#pragma unroll
    for (int t = 0; t < 4; ++t) {
      const _Float16* vp = Vb + (size_t)(t * 16 + ln) * S_ + ks;
      acc[t] = wmma_f16(ap, *(const v16h*)vp, acc[t]);
    }
  }

  // ---- epilogue: normalize (broadcast 1/l into C-layout) and store f16
  const float linv = 1.0f / lrow;
  float invC[8];
#pragma unroll
  for (int r = 0; r < 8; ++r) invC[r] = __shfl(linv, r + g * 8, 32);

  const int b = bh / H_;
  const int h = bh % H_;
#pragma unroll
  for (int r = 0; r < 8; ++r) {
    const int s = q0 + r + g * 8;
#pragma unroll
    for (int t = 0; t < 4; ++t) {
      O16[(size_t)(b * S_ + s) * D_ + h * DK_ + t * 16 + ln] =
          (_Float16)(acc[t][r] * invC[r]);
    }
  }
}

// ---------------------------------------------------------------------------
// Host launcher
// ---------------------------------------------------------------------------
extern "C" void kernel_launch(void* const* d_in, const int* in_sizes, int n_in,
                              void* d_out, int out_size, void* d_ws,
                              size_t ws_size, hipStream_t stream) {
  const float* q = (const float*)d_in[0];
  const float* k = (const float*)d_in[1];
  const float* v = (const float*)d_in[2];
  // d_in[3]: causal mask (int) -- applied analytically in attn_flash
  const float* wq = (const float*)d_in[4];
  const float* bq = (const float*)d_in[5];
  const float* wk = (const float*)d_in[6];
  const float* bk = (const float*)d_in[7];
  const float* wv = (const float*)d_in[8];
  const float* bv = (const float*)d_in[9];
  const float* wo = (const float*)d_in[10];
  const float* bo = (const float*)d_in[11];

  const size_t NX = (size_t)M_ * D_;  // 3,145,728 elems
  const size_t NW = (size_t)D_ * D_;  // 589,824 elems
  _Float16* ws = (_Float16*)d_ws;
  _Float16* q16 = ws;
  _Float16* k16 = q16 + NX;
  _Float16* v16 = k16 + NX;
  _Float16* wq16 = v16 + NX;
  _Float16* wk16 = wq16 + NW;
  _Float16* wv16 = wk16 + NW;
  _Float16* wo16 = wv16 + NW;
  _Float16* Qp = wo16 + NW;   // [B,H,S,DK]
  _Float16* Kp = Qp + NX;     // [B,H,S,DK]
  _Float16* Vt = Kp + NX;     // [B,H,DK,S]
  _Float16* O16 = Vt + NX;    // [B,S,D]

  auto cast = [&](const float* s, _Float16* d, size_t n) {
    const int n4 = (int)(n / 4);
    cast_f32_to_f16<<<(n4 + 255) / 256, 256, 0, stream>>>(s, d, n4);
  };
  cast(q, q16, NX);
  cast(k, k16, NX);
  cast(v, v16, NX);
  cast(wq, wq16, NW);
  cast(wk, wk16, NW);
  cast(wv, wv16, NW);
  cast(wo, wo16, NW);

  // (M/32)*(N/64) = 128*12 = 1536 wave-tiles; 8 waves/block -> 192 blocks
  const dim3 gthr(256);
  const float qscale = 0.125f;  // 1/sqrt(DK) folded into Q
  gemm_wmma16<0><<<192, gthr, 0, stream>>>(q16, wq16, bq, Qp, M_, D_, D_, qscale);
  gemm_wmma16<0><<<192, gthr, 0, stream>>>(k16, wk16, bk, Kp, M_, D_, D_, 1.0f);
  gemm_wmma16<1><<<192, gthr, 0, stream>>>(v16, wv16, bv, Vt, M_, D_, D_, 1.0f);

  // B*H*(S/16) = 3072 wave-tiles -> 384 blocks of 8 waves
  attn_flash<<<384, gthr, 0, stream>>>(Qp, Kp, Vt, O16);

  gemm_wmma16<2><<<192, gthr, 0, stream>>>(O16, wo16, bo, d_out, M_, D_, D_, 1.0f);

  (void)in_sizes;
  (void)n_in;
  (void)out_size;
  (void)ws_size;
}
